// NeuralLogicRec_18485539242365
// MI455X (gfx1250) — compile-verified, matrix operand order checked by reading, not run on hardware
//
#include <hip/hip_runtime.h>
#include <hip/hip_bf16.h>
#include <math.h>

// ---------------------------------------------------------------------------
// NeuralLogicRec fused inference for MI455X (gfx1250, wave32, WMMA bf16)
//
// Shapes: U=100000, I=20000, D=64, H=128, B=64
// Outputs (flat, concat): likes(64x20000) | sim(64x64) | rated(64x20000) | popular(20000)
// ---------------------------------------------------------------------------

#define NUM_ITEMS   20000
#define NUM_B       64
#define DD          64
#define HH          128

#define OFF_LIKES   0
#define OFF_SIM     (NUM_B * NUM_ITEMS)                 // 1,280,000
#define OFF_RATED   (OFF_SIM + NUM_B * NUM_B)           // 1,284,096
#define OFF_POP     (OFF_RATED + NUM_B * NUM_ITEMS)     // 2,564,096

typedef __attribute__((ext_vector_type(16))) __bf16 v16bf;
typedef __attribute__((ext_vector_type(8)))  float  v8f;

// K index for 16-bit WMMA A/B fragment element e (0..15) on lane `ln` (0..31),
// per CDNA5 ISA "16-bit A-Matrix 16x32" layout.
__device__ __forceinline__ int kmap(int ln, int e) {
    if (ln < 16) return (e < 8) ? e       : e + 8;   // K 0..7, 16..23
    else         return (e < 8) ? e + 8   : e + 16;  // K 8..15, 24..31
}

// ---------------------------------------------------------------------------
// Kernel A: user-side precompute (1 block x 256 threads)
//   eu = gather(user_embedding, users)                (64 x 64)
//   Au_l = eu @ likes_W0[:64] + likes_b0              (64 x 128)
//   Au_r = eu @ rated_W0[:64] + rated_b0              (64 x 128)
//   sim  = -cosine(eu, eu)                            (64 x 64) -> d_out
// ---------------------------------------------------------------------------
__global__ __launch_bounds__(256) void user_precompute_kernel(
    const int*   __restrict__ users,
    const float* __restrict__ uemb,
    const float* __restrict__ W0_l, const float* __restrict__ b0_l,
    const float* __restrict__ W0_r, const float* __restrict__ b0_r,
    float* __restrict__ Au_l, float* __restrict__ Au_r,
    float* __restrict__ sim_out)
{
    __shared__ float sEu[NUM_B * DD];
    __shared__ float sNorm[NUM_B];
    const int tid = threadIdx.x;

    for (int idx = tid; idx < NUM_B * DD; idx += 256) {
        int b = idx >> 6, d = idx & 63;
        sEu[idx] = uemb[(long long)users[b] * DD + d];
    }
    __syncthreads();

    for (int idx = tid; idx < NUM_B * HH; idx += 256) {
        int b = idx >> 7, j = idx & 127;
        float sl = 0.f, sr = 0.f;
        #pragma unroll 8
        for (int k = 0; k < DD; ++k) {
            float e = sEu[b * DD + k];
            sl = fmaf(e, W0_l[k * HH + j], sl);
            sr = fmaf(e, W0_r[k * HH + j], sr);
        }
        Au_l[idx] = sl + b0_l[j];
        Au_r[idx] = sr + b0_r[j];
    }

    if (tid < NUM_B) {
        float s = 0.f;
        for (int d = 0; d < DD; ++d) { float e = sEu[tid * DD + d]; s = fmaf(e, e, s); }
        sNorm[tid] = sqrtf(fmaxf(s, 1e-12f));
    }
    __syncthreads();

    for (int idx = tid; idx < NUM_B * NUM_B; idx += 256) {
        int a = idx >> 6, c = idx & 63;
        float s = 0.f;
        #pragma unroll 8
        for (int d = 0; d < DD; ++d) s = fmaf(sEu[a * DD + d], sEu[c * DD + d], s);
        sim_out[idx] = -s / (sNorm[a] * sNorm[c]);
    }
}

// ---------------------------------------------------------------------------
// Kernel B: item-side precompute + popular head (2500 blocks x 128 threads,
// 8 items each).
// ---------------------------------------------------------------------------
__global__ __launch_bounds__(128) void item_precompute_kernel(
    const float* __restrict__ iemb,
    const float* __restrict__ W0_l, const float* __restrict__ W0_r,
    const float* __restrict__ pW0, const float* __restrict__ pb0,
    const float* __restrict__ pW1, const float* __restrict__ pb1,
    const float* __restrict__ pW2, const float* __restrict__ pb2,
    float* __restrict__ Ai_l, float* __restrict__ Ai_r,
    float* __restrict__ popular)
{
    __shared__ float sEi[8 * DD];
    __shared__ float sH0[HH];
    __shared__ float sH1[HH];
    const int tid  = threadIdx.x;          // tid == hidden index j (0..127)
    const int base = blockIdx.x * 8;

    for (int idx = tid; idx < 8 * DD; idx += 128)
        sEi[idx] = iemb[(long long)base * DD + idx];
    __syncthreads();

    for (int il = 0; il < 8; ++il) {
        float sl = 0.f, sr = 0.f, sp = 0.f;
        #pragma unroll 8
        for (int k = 0; k < DD; ++k) {
            float e = sEi[il * DD + k];
            sl = fmaf(e, W0_l[(DD + k) * HH + tid], sl);
            sr = fmaf(e, W0_r[(DD + k) * HH + tid], sr);
            sp = fmaf(e, pW0[k * HH + tid], sp);
        }
        Ai_l[(long long)(base + il) * HH + tid] = sl;
        Ai_r[(long long)(base + il) * HH + tid] = sr;
        sH0[tid] = fmaxf(sp + pb0[tid], 0.f);
        __syncthreads();

        float s1 = 0.f;
        #pragma unroll 8
        for (int k = 0; k < HH; ++k) s1 = fmaf(sH0[k], pW1[k * HH + tid], s1);
        sH1[tid] = fmaxf(s1 + pb1[tid], 0.f) * pW2[tid];
        __syncthreads();

        if (tid == 0) {
            float tot = 0.f;
            for (int k = 0; k < HH; ++k) tot += sH1[k];
            tot += pb2[0];
            popular[base + il] = 1.f / (1.f + __expf(-tot));
        }
        __syncthreads();
    }
}

// ---------------------------------------------------------------------------
// Kernel C: one-shot W1 -> packed bf16 B-fragment layout in workspace.
// Fragment (ks,nt): 32 lanes x 8 dwords; dword e2 of lane ln holds
// K = ks*32 + kmap(ln, 2*e2) .. +1, N = nt*16 + (ln & 15).
// grid (32, 2): y selects head.
// ---------------------------------------------------------------------------
__global__ __launch_bounds__(256) void pack_w1_kernel(
    const float* __restrict__ W1_l, const float* __restrict__ W1_r,
    unsigned int* __restrict__ dst_l, unsigned int* __restrict__ dst_r)
{
    const float* W1        = blockIdx.y ? W1_r  : W1_l;
    unsigned int* dst      = blockIdx.y ? dst_r : dst_l;
    int idxU = blockIdx.x * 256 + threadIdx.x;   // 0..8191
    int e2 = idxU & 7;
    int ln = (idxU >> 3) & 31;
    int nt = (idxU >> 8) & 7;
    int ks = idxU >> 11;
    int K0 = ks * 32 + kmap(ln, 2 * e2);
    int N  = nt * 16 + (ln & 15);
    union { __bf16 h[2]; unsigned int u; } c;
    c.h[0] = (__bf16)W1[K0 * HH + N];        // native v_cvt, RNE
    c.h[1] = (__bf16)W1[(K0 + 1) * HH + N];
    dst[idxU] = c.u;
}

// ---------------------------------------------------------------------------
// Main fused head kernel: grid (1250 item-blocks, 4 user-blocks, 2 heads),
// 256 threads = 8 wave32.  Each wave: TWO items x 16 users; each B fragment
// load feeds two v_wmma_f32_16x16x32_bf16.
//   h0 = relu(Au[16x128] + Ai[i])  (built in bf16 A-fragment layout)
//   h1 = h0 @ W1                   (64x wmma per wave, fp32 acc)
//   out[16 users] = sigmoid(relu(h1 + b1) @ W2 + b2)
// ---------------------------------------------------------------------------
__global__ __launch_bounds__(256) void fused_head_kernel(
    const float* __restrict__ Au_l, const float* __restrict__ Au_r,
    const float* __restrict__ Ai_l, const float* __restrict__ Ai_r,
    const unsigned int* __restrict__ W1p_l, const unsigned int* __restrict__ W1p_r,
    const float* __restrict__ b1_l, const float* __restrict__ W2_l, const float* __restrict__ b2_l,
    const float* __restrict__ b1_r, const float* __restrict__ W2_r, const float* __restrict__ b2_r,
    float* __restrict__ out)
{
    const int head = blockIdx.z;
    const float* Au        = head ? Au_r  : Au_l;
    const float* Ai        = head ? Ai_r  : Ai_l;
    const unsigned int* Wp = head ? W1p_r : W1p_l;
    const float* B1        = head ? b1_r  : b1_l;
    const float* W2        = head ? W2_r  : W2_l;
    const float* B2        = head ? b2_r  : b2_l;
    float* o = out + (head ? OFF_RATED : OFF_LIKES);

    const int ib   = blockIdx.x;     // item block: 16 items
    const int ub   = blockIdx.y;     // user block: 16 users
    const int tid  = threadIdx.x;
    const int lane = tid & 31;
    const int wave = tid >> 5;

    __shared__ float sAu[16 * HH];                       // 8 KB
    __shared__ float sAi[16 * HH];                       // 8 KB
    __shared__ __align__(16) unsigned int sW1[8192];     // 32 KB, packed B frags
    __shared__ float sB1[HH];
    __shared__ float sW2[HH];

    for (int idx = tid; idx < 16 * HH; idx += 256)
        sAu[idx] = Au[(ub * 16) * HH + idx];
    for (int idx = tid; idx < 16 * HH; idx += 256)
        sAi[idx] = Ai[(long long)(ib * 16) * HH + idx];
    if (tid < HH) { sB1[tid] = B1[tid]; sW2[tid] = W2[tid]; }
    // copy pre-packed W1 fragments (b128 global loads -> b128 LDS stores)
    {
        uint4* d = (uint4*)sW1;
        const uint4* s = (const uint4*)Wp;
        for (int idx = tid; idx < 2048; idx += 256) d[idx] = s[idx];
    }
    __syncthreads();

    const int il0 = wave * 2;        // first item of this wave
    const int m   = lane & 15;       // user row within tile

    v8f acc[2][8];
    #pragma unroll
    for (int j = 0; j < 2; ++j)
        #pragma unroll
        for (int nt = 0; nt < 8; ++nt)
            #pragma unroll
            for (int v = 0; v < 8; ++v) acc[j][nt][v] = 0.f;

    #pragma unroll
    for (int ks = 0; ks < 4; ++ks) {
        // Build A fragments for both items: h0 = relu(Au[m] + Ai[il]), bf16.
        v16bf af[2];
        #pragma unroll
        for (int j = 0; j < 2; ++j) {
            #pragma unroll
            for (int p = 0; p < 8; ++p) {
                int K0 = ks * 32 + kmap(lane, 2 * p);
                float a0 = fmaxf(sAu[m * HH + K0]     + sAi[(il0 + j) * HH + K0],     0.f);
                float a1 = fmaxf(sAu[m * HH + K0 + 1] + sAi[(il0 + j) * HH + K0 + 1], 0.f);
                af[j][2 * p]     = (__bf16)a0;   // native v_cvt, RNE
                af[j][2 * p + 1] = (__bf16)a1;
            }
        }
        #pragma unroll
        for (int nt = 0; nt < 8; ++nt) {
            union { unsigned int u[8]; v16bf v; } wf;
            const uint4* pW = (const uint4*)&sW1[((ks * 8 + nt) * 32 + lane) * 8];
            uint4 w0 = pW[0], w1 = pW[1];
            wf.u[0] = w0.x; wf.u[1] = w0.y; wf.u[2] = w0.z; wf.u[3] = w0.w;
            wf.u[4] = w1.x; wf.u[5] = w1.y; wf.u[6] = w1.z; wf.u[7] = w1.w;
            acc[0][nt] = __builtin_amdgcn_wmma_f32_16x16x32_bf16(
                false, af[0], false, wf.v, (short)0, acc[0][nt], false, false);
            acc[1][nt] = __builtin_amdgcn_wmma_f32_16x16x32_bf16(
                false, af[1], false, wf.v, (short)0, acc[1][nt], false, false);
        }
    }

    // Epilogue: layer-2 dot on accumulator layout.
    // acc[j][nt], VGPR v, lane l -> element (M = v + 8*(l>=16), N = nt*16 + l%16)
    const int col  = lane & 15;
    const int half = lane >> 4;
    const float b2v = B2[0];
    #pragma unroll
    for (int j = 0; j < 2; ++j) {
        float s[8];
        #pragma unroll
        for (int v = 0; v < 8; ++v) s[v] = 0.f;
        #pragma unroll
        for (int nt = 0; nt < 8; ++nt) {
            int n = nt * 16 + col;
            float b1v = sB1[n], w2v = sW2[n];
            #pragma unroll
            for (int v = 0; v < 8; ++v)
                s[v] = fmaf(fmaxf(acc[j][nt][v] + b1v, 0.f), w2v, s[v]);
        }
        // reduce across the 16 lanes of each half (xor masks 1,2,4,8 stay in-half)
        #pragma unroll
        for (int v = 0; v < 8; ++v)
            for (int off = 1; off < 16; off <<= 1)
                s[v] += __shfl_xor(s[v], off, 32);

        const int item = ib * 16 + il0 + j;
        if (col == 0) {
            #pragma unroll
            for (int v = 0; v < 8; ++v) {
                int row = ub * 16 + half * 8 + v;
                float x = s[v] + b2v;
                o[(long long)row * NUM_ITEMS + item] = 1.f / (1.f + __expf(-x));
            }
        }
    }
}

// ---------------------------------------------------------------------------
extern "C" void kernel_launch(void* const* d_in, const int* in_sizes, int n_in,
                              void* d_out, int out_size, void* d_ws, size_t ws_size,
                              hipStream_t stream) {
    const int*   users = (const int*)d_in[0];
    const float* uemb  = (const float*)d_in[1];
    const float* iemb  = (const float*)d_in[2];
    const float* lW0 = (const float*)d_in[3],  *lb0 = (const float*)d_in[4];
    const float* lW1 = (const float*)d_in[5],  *lb1 = (const float*)d_in[6];
    const float* lW2 = (const float*)d_in[7],  *lb2 = (const float*)d_in[8];
    const float* rW0 = (const float*)d_in[9],  *rb0 = (const float*)d_in[10];
    const float* rW1 = (const float*)d_in[11], *rb1 = (const float*)d_in[12];
    const float* rW2 = (const float*)d_in[13], *rb2 = (const float*)d_in[14];
    const float* pW0 = (const float*)d_in[15], *pb0 = (const float*)d_in[16];
    const float* pW1 = (const float*)d_in[17], *pb1 = (const float*)d_in[18];
    const float* pW2 = (const float*)d_in[19], *pb2 = (const float*)d_in[20];

    float* out = (float*)d_out;
    float* ws  = (float*)d_ws;
    // workspace layout (floats):
    //   Au_l[8192] | Au_r[8192] | Ai_l[2.56M] | Ai_r[2.56M] | W1p_l[8192] | W1p_r[8192]
    float* Au_l = ws;
    float* Au_r = ws + NUM_B * HH;
    float* Ai_l = ws + 2 * NUM_B * HH;
    float* Ai_r = Ai_l + (size_t)NUM_ITEMS * HH;
    unsigned int* W1p_l = (unsigned int*)(Ai_r + (size_t)NUM_ITEMS * HH);
    unsigned int* W1p_r = W1p_l + 8192;

    float* sim     = out + OFF_SIM;
    float* popular = out + OFF_POP;

    user_precompute_kernel<<<1, 256, 0, stream>>>(
        users, uemb, lW0, lb0, rW0, rb0, Au_l, Au_r, sim);

    pack_w1_kernel<<<dim3(32, 2), 256, 0, stream>>>(lW1, rW1, W1p_l, W1p_r);

    item_precompute_kernel<<<NUM_ITEMS / 8, 128, 0, stream>>>(
        iemb, lW0, rW0, pW0, pb0, pW1, pb1, pW2, pb2, Ai_l, Ai_r, popular);

    fused_head_kernel<<<dim3(NUM_ITEMS / 16, NUM_B / 16, 2), 256, 0, stream>>>(
        Au_l, Au_r, Ai_l, Ai_r, W1p_l, W1p_r,
        lb1, lW2, lb2, rb1, rW2, rb2,
        out);
}
